// cRNN_62569083568454
// MI455X (gfx1250) — matrix operand order repaired and checked by reading, not compiled, not run on hardware
//
#include <hip/hip_runtime.h>
#include <math.h>

// ---------------------------------------------------------------------------
// cRNN scan + BN + head for MI455X (gfx1250, wave32, WMMA)
//   k1: persistent-batch-tile scan; input projection FUSED into the WMMA
//       (A rows K=100..103 carry x_t, B rows carry W_in), 4 chained
//       v_wmma_f32_16x16x32_bf16 per step, one barrier per step,
//       hardware v_tanh_f32, NT streaming, IOFFSET-folded addressing
//   k2: BN stats reduce (deterministic, no atomics)
//   k3: normalize + tanh + fc head (wave-per-position, shfl reduce)
// ---------------------------------------------------------------------------

typedef __bf16 bf16x16 __attribute__((ext_vector_type(16)));
typedef float  f32x8   __attribute__((ext_vector_type(8)));

#define T_STEPS 2000
#define HID     100
#define NPAD    112            // 7 WMMA n-tiles of 16
#define K2N     64             // packed bf16 K-pairs (K padded 100(+4 x rows) -> 128)
#define C1f     0.9f
#define C2f     0.1f
#define BT      512000         // B*T

// workspace layout (float offsets)
#define WS_SUM_OFF   0         // 16 WGs x 128 channel sums
#define WS_SQ_OFF    2048      // 16 WGs x 128 channel sumsq
#define WS_SCALE_OFF 4096      // 128
#define WS_BIAS_OFF  4224      // 128
#define WS_HALL_OFF  8192      // hidden trajectory [B][T][H] f32 (204.8 MB)

__device__ __forceinline__ unsigned short f2bf(float f) {
  unsigned int u = __float_as_uint(f);
  u += 0x7FFFu + ((u >> 16) & 1u);     // round-to-nearest-even
  return (unsigned short)(u >> 16);
}

// CDNA5 hardware V_TANH_F32 transcendental (confirmed lowering last round).
__device__ __forceinline__ float fast_tanh(float x) {
#if __has_builtin(__builtin_amdgcn_tanhf)
  return __builtin_amdgcn_tanhf(x);
#elif __has_builtin(__builtin_amdgcn_tanh_f32)
  return __builtin_amdgcn_tanh_f32(x);
#else
  return tanhf(x);
#endif
}

// ---------------------------------------------------------------------------
// Kernel 1: the scan. grid = 16 (batch tiles of 16 rows), block = 224 (7 waves)
// ---------------------------------------------------------------------------
__global__ __launch_bounds__(224) void rnn_scan_kernel(
    const float* __restrict__ x,      // [B,T,4]
    const float* __restrict__ W_in,   // [4,H]
    const float* __restrict__ W_hid,  // [H,H]
    const float* __restrict__ sigma,  // [B,T,H]
    float* __restrict__ h_last_out,   // [B,H]
    float* __restrict__ ws)
{
  __shared__ unsigned int Wpk[K2N * NPAD];      // bf16 pairs: K<100 W_hid, K=100..103 W_in
  __shared__ unsigned int Apk[2][K2N * 16];     // double-buffered A: tanh(h) rows + x rows

  const int tid  = (int)threadIdx.x;
  const int wave = tid >> 5;
  const int lane = tid & 31;
  const int half = lane >> 4;
  const int l15  = lane & 15;
  const int n    = wave * 16 + l15;        // this thread's output column
  const int b0   = (int)blockIdx.x * 16;   // batch tile base
  const bool act = (n < HID);

  // --- one-time staging: B = [W_hidden ; W_in ; 0] as packed bf16 [k2][n]
  for (int idx = tid; idx < K2N * NPAD; idx += 224) {
    int k2 = idx / NPAD, nn = idx % NPAD;
    int k0 = 2 * k2, k1 = k0 + 1;
    unsigned int lo = 0u, hi = 0u;
    if (nn < HID) {
      if (k0 < HID)            lo = f2bf(W_hid[k0 * HID + nn]);
      else if (k0 < HID + 4)   lo = f2bf(W_in[(k0 - HID) * HID + nn]);
      if (k1 < HID)            hi = f2bf(W_hid[k1 * HID + nn]);
      else if (k1 < HID + 4)   hi = f2bf(W_in[(k1 - HID) * HID + nn]);
    }
    Wpk[idx] = lo | (hi << 16);
  }
  // zero both A buffers (K >= 104 rows stay zero forever)
  for (int idx = tid; idx < K2N * 16; idx += 224) {
    Apk[0][idx] = 0u;
    Apk[1][idx] = 0u;
  }
  __syncthreads();

  // --- loop-invariant B fragments: 4 K-tiles x 8 dwords, resident in VGPRs
  union ABu { unsigned int u[8]; bf16x16 v; };
  ABu bfr[4];
#pragma unroll
  for (int kt = 0; kt < 4; ++kt)
#pragma unroll
    for (int v = 0; v < 8; ++v)
      bfr[kt].u[v] = Wpk[(kt * 16 + ((v < 4) ? (4 * half + v) : (8 + 4 * half + (v - 4)))) * NPAD + n];

  // --- per-thread resident state: rows M = r + 8*half of column n
  float hreg[8], sum[8], sq[8];
  unsigned short abits[8];               // bf16(tanh(h)) for the NEXT publish
#pragma unroll
  for (int r = 0; r < 8; ++r) { hreg[r] = 0.f; sum[r] = 0.f; sq[r] = 0.f; abits[r] = 0; }

  // per-thread streaming row base: r-term (r*T*H*4 = 800000B) fits in IOFFSET
  const size_t rowBase = (size_t)(b0 + 8 * half) * T_STEPS * HID + n;
  const float* sigP = sigma + rowBase;
  float*       hP   = ws + WS_HALL_OFF + rowBase;

  // x staging: tid<64 owns (m = tid>>2, i = tid&3); preload t=0 into a register
  const int xm = tid >> 2, xi = tid & 3;
  const float* xP = x + (size_t)(b0 + xm) * T_STEPS * 4 + xi;
  const int xAddr = ((50 + (xi >> 1)) * 16 + xm) * 2 + (xi & 1);  // A row K=100+xi
  float xnext = (tid < 64) ? xP[0] : 0.f;

  for (int t = 0; t < T_STEPS; ++t) {
    const int buf = t & 1;
    unsigned short* Abf = (unsigned short*)Apk[buf];

    // phase 1: publish pre-computed tanh bits + x_t bits (short critical section)
    if (act) {
#pragma unroll
      for (int r = 0; r < 8; ++r) {
        int M = r + 8 * half;
        Abf[((n >> 1) * 16 + M) * 2 + (n & 1)] = abits[r];
      }
    }
    if (tid < 64) Abf[xAddr] = f2bf(xnext);
    __syncthreads();   // the ONLY barrier per step (double buffering covers WAR)

    // phase 2: 4 chained bf16 WMMAs -> c = tanh(h)@W_hid + x_t@W_in  (f32 acc)
    f32x8 c;
#pragma unroll
    for (int i = 0; i < 8; ++i) c[i] = 0.f;
#pragma unroll
    for (int kt = 0; kt < 4; ++kt) {
      ABu af;
#pragma unroll
      for (int v = 0; v < 8; ++v) {
        int k2 = 16 * kt + ((v < 4) ? (4 * half + v) : (8 + 4 * half + (v - 4)));
        af.u[v] = Apk[buf][k2 * 16 + l15];
      }
      c = __builtin_amdgcn_wmma_f32_16x16x32_bf16(false, af.v, false, bfr[kt].v,
                                                  (short)0, c, false, false);
    }

    // phase 3: h update, trajectory NT-store, stats, next-step tanh + x load
    const size_t toff = (size_t)t * HID;
    if (act) {
#pragma unroll
      for (int r = 0; r < 8; ++r) {
        size_t off = toff + (size_t)r * (T_STEPS * HID);
        float s  = __builtin_nontemporal_load(sigP + off);
        float hn = C1f * hreg[r] + C2f * c[r] + s;
        hreg[r] = hn;
        __builtin_nontemporal_store(hn, hP + off);
        sum[r] += hn;
        sq[r]  += hn * hn;
        abits[r] = f2bf(fast_tanh(hn));  // single V_TANH_F32, off critical section
      }
      if (t + 1 < T_STEPS)
        __builtin_prefetch(sigP + toff + HID, 0, 0);
    }
    if (tid < 64 && t + 1 < T_STEPS)
      xnext = xP[(size_t)(t + 1) * 4];   // overlapped with next barrier approach
  }

  // --- deterministic per-WG channel partials (reduce over M, no atomics)
  float sA = 0.f, sQ = 0.f;
#pragma unroll
  for (int r = 0; r < 8; ++r) { sA += sum[r]; sQ += sq[r]; }
  sA += __shfl_xor(sA, 16, 32);
  sQ += __shfl_xor(sQ, 16, 32);
  if (lane < 16 && act) {
    ws[WS_SUM_OFF + (int)blockIdx.x * 128 + n] = sA;
    ws[WS_SQ_OFF  + (int)blockIdx.x * 128 + n] = sQ;
  }
  // --- h_last straight from registers
  if (act) {
#pragma unroll
    for (int r = 0; r < 8; ++r) {
      int M = r + 8 * half;
      h_last_out[(size_t)(b0 + M) * HID + n] = hreg[r];
    }
  }
}

// ---------------------------------------------------------------------------
// Kernel 2: fold per-WG partials into BN scale/bias
// ---------------------------------------------------------------------------
__global__ void stats_kernel(const float* __restrict__ gamma,
                             const float* __restrict__ beta,
                             float* __restrict__ ws) {
  int nn = (int)threadIdx.x;  // 0..127
  if (nn < HID) {
    float s = 0.f, q = 0.f;
    for (int g = 0; g < 16; ++g) {
      s += ws[WS_SUM_OFF + g * 128 + nn];
      q += ws[WS_SQ_OFF  + g * 128 + nn];
    }
    float inv  = 1.0f / (float)BT;
    float mean = s * inv;
    float var  = fmaxf(q * inv - mean * mean, 0.f);
    float scale = rsqrtf(var + 1e-5f) * gamma[nn];
    ws[WS_SCALE_OFF + nn] = scale;
    ws[WS_BIAS_OFF  + nn] = beta[nn] - mean * scale;
  } else {
    ws[WS_SCALE_OFF + nn] = 0.f;
    ws[WS_BIAS_OFF  + nn] = 0.f;
  }
}

// ---------------------------------------------------------------------------
// Kernel 3: normalize + tanh + fc head. One wave per (b,t) position.
// ---------------------------------------------------------------------------
__global__ __launch_bounds__(256) void head_kernel(
    const float* __restrict__ fc_w, const float* __restrict__ fc_b,
    const float* __restrict__ ws, float* __restrict__ out) {
  __shared__ float sS[128], sB[128], sW[128];
  int tid = (int)threadIdx.x, lane = tid & 31, wv = tid >> 5;
  if (tid < 128) {
    sS[tid] = ws[WS_SCALE_OFF + tid];
    sB[tid] = ws[WS_BIAS_OFF + tid];
    sW[tid] = (tid < HID) ? fc_w[tid] : 0.f;
  }
  __syncthreads();
  const float* h_all = ws + WS_HALL_OFF;
  const float fcb = fc_b[0];
  int nwaves = (int)gridDim.x * 8;
  for (int pos = (int)blockIdx.x * 8 + wv; pos < BT; pos += nwaves) {
    float acc = 0.f;
#pragma unroll
    for (int k = 0; k < 4; ++k) {
      int nn = lane + 32 * k;
      if (nn < HID) {
        float hv = __builtin_nontemporal_load(&h_all[(size_t)pos * HID + nn]);
        acc += fast_tanh(hv * sS[nn] + sB[nn]) * sW[nn];
      }
    }
#pragma unroll
    for (int m = 16; m >= 1; m >>= 1) acc += __shfl_xor(acc, m, 32);
    if (lane == 0) out[pos] = acc + fcb;
  }
}

// ---------------------------------------------------------------------------
extern "C" void kernel_launch(void* const* d_in, const int* in_sizes, int n_in,
                              void* d_out, int out_size, void* d_ws, size_t ws_size,
                              hipStream_t stream) {
  const float* x     = (const float*)d_in[0];
  const float* W_in  = (const float*)d_in[1];
  const float* W_hid = (const float*)d_in[2];
  const float* sigma = (const float*)d_in[3];
  const float* gamma = (const float*)d_in[4];
  const float* beta  = (const float*)d_in[5];
  const float* fc_w  = (const float*)d_in[6];
  const float* fc_b  = (const float*)d_in[7];
  float* out = (float*)d_out;            // [B*T] out, then [B*H] h_last
  float* ws  = (float*)d_ws;             // needs ~204.9 MB

  rnn_scan_kernel<<<16, 224, 0, stream>>>(x, W_in, W_hid, sigma, out + BT, ws);
  stats_kernel<<<1, 128, 0, stream>>>(gamma, beta, ws);
  head_kernel<<<2048, 256, 0, stream>>>(fc_w, fc_b, ws, out);
}
